// UniMolModel_66468913873366
// MI455X (gfx1250) — compile-verified, matrix operand order checked by reading, not run on hardware
//
#include <hip/hip_runtime.h>
#include <math.h>

#define BB 8
#define LL 256
#define DD 512
#define HH 64
#define DHH 8
#define KK 128
#define FF 2048
#define NLAYER 15
#define MROWS (BB * LL)          // 2048 token rows
#define NPAIR (BB * LL * LL)     // 524288 pair rows

typedef __attribute__((ext_vector_type(16))) _Float16 v16h;
typedef __attribute__((ext_vector_type(8)))  _Float16 v8h;
typedef __attribute__((ext_vector_type(2)))  _Float16 v2h;
typedef __attribute__((ext_vector_type(8)))  float    v8f;

// ---------------------------------------------------------------------------
// helpers
// ---------------------------------------------------------------------------
__device__ __forceinline__ float gelu_exact(float x) {
    return 0.5f * x * (1.0f + erff(x * 0.70710678118654752f));
}

__device__ __forceinline__ float wave_sum(float v) {
#pragma unroll
    for (int m = 16; m > 0; m >>= 1) v += __shfl_xor(v, m, 32);
    return v;
}
__device__ __forceinline__ float wave_max(float v) {
#pragma unroll
    for (int m = 16; m > 0; m >>= 1) v = fmaxf(v, __shfl_xor(v, m, 32));
    return v;
}

// Assemble a 16-bit WMMA A/B fragment from an LDS row (row-of-K layout).
// Per ISA 7.12.2: lanes 0-15 hold K in {0..7, 16..23}, lanes 16-31 hold
// K in {8..15, 24..31}; caller passes koff = (lane>>4)*8.
__device__ __forceinline__ v16h frag_ld(const _Float16* rowbase, int koff) {
    v8h lo = *(const v8h*)(rowbase + koff);
    v8h hi = *(const v8h*)(rowbase + koff + 16);
    v16h r;
#pragma unroll
    for (int i = 0; i < 8; ++i) { r[i] = lo[i]; r[i + 8] = hi[i]; }
    return r;
}

// ---------------------------------------------------------------------------
// Fused pair-bias head: Gaussian basis -> 128x128 GEMM -> GELU -> 128x64 GEMM
// -> transpose-scatter into bias(B,H,L,L) with -inf at padded key columns.
// Block = 128 threads = 4 waves; each wave owns one 16-row pair tile.
// All GEMMs via v_wmma_f32_16x16x32_f16.
// ---------------------------------------------------------------------------
union GbfSh {
    _Float16 w1[KK * KK];       // proj_w1 staged N-major: w1[n*128 + k]
    _Float16 a2[4][16 * KK];    // per-wave GELU output (layer-2 A tiles)
};

__launch_bounds__(128)
__global__ void gbf_bias_kernel(const int* __restrict__ tokens,
                                const float* __restrict__ dist,
                                const int* __restrict__ etype,
                                const float* __restrict__ means,
                                const float* __restrict__ stds,
                                const float* __restrict__ gmul,
                                const float* __restrict__ gbias,
                                const float* __restrict__ w1,
                                const float* __restrict__ b1,
                                const float* __restrict__ w2,
                                const float* __restrict__ b2,
                                float* __restrict__ biasout)
{
    __shared__ __align__(16) GbfSh sh;
    __shared__ __align__(16) _Float16 sw2[HH * KK]; // proj_w2 N-major: [h][k]
    __shared__ float sm[KK], sinv[KK], scoef[KK];

    const int t = threadIdx.x;
    // stage weights + gaussian params (fp32 -> f16, transposed to N-major)
#pragma unroll
    for (int it = 0; it < KK; ++it) {               // 128 elems per thread
        int idx = t + it * 128;                     // w1 is [k][n]
        int kk = idx >> 7, nn = idx & 127;
        sh.w1[nn * KK + kk] = (_Float16)w1[idx];
    }
#pragma unroll
    for (int it = 0; it < 64; ++it) {
        int idx = t + it * 128;                     // w2 is [k][h]
        int kk = idx >> 6, hh = idx & 63;
        sw2[hh * KK + kk] = (_Float16)w2[idx];
    }
    {
        float sd = fabsf(stds[t]) + 1e-5f;
        float inv = 1.0f / sd;
        sm[t] = means[t];
        sinv[t] = inv;
        scoef[t] = 0.398942280401f * inv;           // 1/(sqrt(2*3.14159)*sd)
    }
    __syncthreads();

    const int lane = t & 31, wid = t >> 5;
    const int m0 = (blockIdx.x * 4 + wid) * 16;     // first pair row of tile
    const int row = lane & 15;
    const int koff = (lane >> 4) * 8;
    const int nl = lane & 15;

    // per-lane scalar affine of distance (rows duplicated in lane, lane+16)
    const int p = m0 + row;
    const int e = etype[p];
    const float y = gmul[e] * dist[p] + gbias[e];

    // Gaussian basis straight into WMMA A fragments (K = 128 -> 4 k-steps)
    v16h af[4];
#pragma unroll
    for (int ks = 0; ks < 4; ++ks) {
#pragma unroll
        for (int u = 0; u < 8; ++u) {
            int k1 = ks * 32 + koff + u;
            int k2 = k1 + 16;
            float a1 = (y - sm[k1]) * sinv[k1];
            float a2v = (y - sm[k2]) * sinv[k2];
            af[ks][u]     = (_Float16)(__expf(-0.5f * a1 * a1) * scoef[k1]);
            af[ks][u + 8] = (_Float16)(__expf(-0.5f * a2v * a2v) * scoef[k2]);
        }
    }

    // Layer 1: 16x128 = 8 N-tiles, K=128; results -> registers (GELU'd f16)
    _Float16 g16[64];
#pragma unroll
    for (int nt = 0; nt < 8; ++nt) {
        v8f acc = {};
#pragma unroll
        for (int ks = 0; ks < 4; ++ks) {
            v16h bf = frag_ld(sh.w1 + (nt * 16 + nl) * KK + ks * 32, koff);
            acc = __builtin_amdgcn_wmma_f32_16x16x32_f16(
                false, af[ks], false, bf, (short)0, acc, false, false);
        }
        float bn = b1[nt * 16 + nl];
#pragma unroll
        for (int ee = 0; ee < 8; ++ee)
            g16[nt * 8 + ee] = (_Float16)gelu_exact(acc[ee] + bn);
    }
    __syncthreads();                // all waves done reading sh.w1
    // transpose C-fragments (lane=n) into A-tile layout (lane=m) via LDS
#pragma unroll
    for (int nt = 0; nt < 8; ++nt)
#pragma unroll
        for (int ee = 0; ee < 8; ++ee) {
            int m = (lane >> 4) * 8 + ee;
            sh.a2[wid][m * KK + nt * 16 + nl] = g16[nt * 8 + ee];
        }
    __syncthreads();

    // Layer 2: 16x64 output, K=128
    const _Float16* arow2 = sh.a2[wid] + row * KK;
    v16h a2f[4];
#pragma unroll
    for (int ks = 0; ks < 4; ++ks) a2f[ks] = frag_ld(arow2 + ks * 32, koff);

#pragma unroll
    for (int ht = 0; ht < 4; ++ht) {
        v8f acc = {};
#pragma unroll
        for (int ks = 0; ks < 4; ++ks) {
            v16h bf = frag_ld(sw2 + (ht * 16 + nl) * KK + ks * 32, koff);
            acc = __builtin_amdgcn_wmma_f32_16x16x32_f16(
                false, a2f[ks], false, bf, (short)0, acc, false, false);
        }
        const int h = ht * 16 + nl;
        const float bh = b2[h];
#pragma unroll
        for (int ee = 0; ee < 8; ++ee) {
            int m  = (lane >> 4) * 8 + ee;
            int pp = m0 + m;
            int j  = pp & (LL - 1);
            int i2 = (pp >> 8) & (LL - 1);
            int bb = pp >> 16;
            float val = acc[ee] + bh;
            if (tokens[bb * LL + j] == 0) val = -__builtin_inff();
            biasout[(((size_t)(bb * HH + h) * LL + i2) * LL) + j] = val;
        }
    }
}

// ---------------------------------------------------------------------------
// Generic WMMA GEMM: out = epilogue( (A_f16[MxK] @ W_f32[KxN] + bias) * scale )
// ACT: 0=none 1=gelu. HAS_RES adds fp32 residual. OUT_HALF selects f16/f32 out.
// Block 128 thr = 4 waves; block tile 32x128; wave tile 16x64 (1 A-frag feeds
// 4 WMMAs); double-buffered LDS -> one barrier per 32-wide K step.
// ---------------------------------------------------------------------------
template<int ACT, int HAS_RES, int OUT_HALF>
__launch_bounds__(128)
__global__ void gemm_kernel(const _Float16* __restrict__ A,
                            const float* __restrict__ W,
                            const float* __restrict__ bias,
                            const float* __restrict__ res,
                            void* __restrict__ outp,
                            int M, int N, int K, float scale)
{
    __shared__ __align__(16) _Float16 sA[2][32 * 40];    //  5 KB
    __shared__ __align__(16) _Float16 sW[2][128 * 40];   // 20 KB, N-major

    const int t = threadIdx.x;
    const int lane = t & 31, wid = t >> 5;
    const int wm = wid & 1, wn = wid >> 1;   // 2x2 waves over 32x128
    const int n0 = blockIdx.x * 128;
    const int m0 = blockIdx.y * 32;
    const int nl = lane & 15;
    const int koff = (lane >> 4) * 8;

    v8f acc[4] = {};

    auto stage = [&](int ks, int bufi) {
        // A tile: 32x32 f16, one b128 per thread
        {
            int r = t >> 2, c = (t & 3) * 8;
            v8h av = *(const v8h*)(A + (size_t)(m0 + r) * K + ks + c);
            *(v8h*)(&sA[bufi][r * 40 + c]) = av;
        }
        // W tile: 32x128 fp32 -> f16 N-major; thread t owns column n0+t,
        // converts two K rows per iter into one packed ds_store_b32.
#pragma unroll
        for (int i = 0; i < 16; ++i) {
            int kk = 2 * i;
            float f0 = W[(size_t)(ks + kk) * N + n0 + t];
            float f1 = W[(size_t)(ks + kk + 1) * N + n0 + t];
            v2h pk;
            pk[0] = (_Float16)f0;
            pk[1] = (_Float16)f1;
            *(v2h*)(&sW[bufi][t * 40 + kk]) = pk;
        }
    };

    const int nk = K / 32;
    stage(0, 0);
    __syncthreads();
    for (int s = 0; s < nk; ++s) {
        if (s + 1 < nk) stage((s + 1) * 32, (s + 1) & 1);  // fill other buffer
        const _Float16* a = sA[s & 1];
        const _Float16* w = sW[s & 1];
        v16h afr = frag_ld(a + (wm * 16 + nl) * 40, koff);
#pragma unroll
        for (int f = 0; f < 4; ++f) {
            v16h bf = frag_ld(w + (wn * 64 + f * 16 + nl) * 40, koff);
            acc[f] = __builtin_amdgcn_wmma_f32_16x16x32_f16(
                false, afr, false, bf, (short)0, acc[f], false, false);
        }
        __syncthreads();   // reads of buf s done; writes of buf s+1 visible
    }

    const int mbase = m0 + wm * 16 + (lane >> 4) * 8;
#pragma unroll
    for (int f = 0; f < 4; ++f) {
        int n = n0 + wn * 64 + f * 16 + nl;
        float bv = bias[n];
#pragma unroll
        for (int ee = 0; ee < 8; ++ee) {
            size_t idx = (size_t)(mbase + ee) * N + n;
            float vv = (acc[f][ee] + bv) * scale;
            if (ACT == 1) vv = gelu_exact(vv);
            if (HAS_RES)  vv += res[idx];
            if (OUT_HALF) ((_Float16*)outp)[idx] = (_Float16)vv;
            else          ((float*)outp)[idx] = vv;
        }
    }
}

// ---------------------------------------------------------------------------
// LayerNorm (wave per row) -> f16 activations for WMMA GEMMs
// ---------------------------------------------------------------------------
__launch_bounds__(256)
__global__ void ln_to_half_kernel(const float* __restrict__ x,
                                  const float* __restrict__ g,
                                  const float* __restrict__ beta,
                                  _Float16* __restrict__ out)
{
    const int wid = threadIdx.x >> 5, lane = threadIdx.x & 31;
    const int row = blockIdx.x * 8 + wid;
    const float* xr = x + (size_t)row * DD;
    float v[16], s = 0.f, sq = 0.f;
#pragma unroll
    for (int c = 0; c < 16; ++c) {
        v[c] = xr[lane + 32 * c];
        s += v[c]; sq += v[c] * v[c];
    }
    s = wave_sum(s); sq = wave_sum(sq);
    const float mean = s * (1.0f / DD);
    const float inv  = rsqrtf(sq * (1.0f / DD) - mean * mean + 1e-5f);
#pragma unroll
    for (int c = 0; c < 16; ++c) {
        int col = lane + 32 * c;
        out[(size_t)row * DD + col] =
            (_Float16)((v[c] - mean) * inv * g[col] + beta[col]);
    }
}

// Embedding gather + embed-LN + pad zeroing -> fp32 residual stream x
__launch_bounds__(256)
__global__ void embed_ln_kernel(const int* __restrict__ tokens,
                                const float* __restrict__ emb,
                                const float* __restrict__ g,
                                const float* __restrict__ beta,
                                float* __restrict__ x)
{
    const int wid = threadIdx.x >> 5, lane = threadIdx.x & 31;
    const int row = blockIdx.x * 8 + wid;
    const int tok = tokens[row];
    const float* er = emb + (size_t)tok * DD;
    float v[16], s = 0.f, sq = 0.f;
#pragma unroll
    for (int c = 0; c < 16; ++c) {
        v[c] = er[lane + 32 * c];
        s += v[c]; sq += v[c] * v[c];
    }
    s = wave_sum(s); sq = wave_sum(sq);
    const float mean = s * (1.0f / DD);
    const float inv  = rsqrtf(sq * (1.0f / DD) - mean * mean + 1e-5f);
    const float w = (tok == 0) ? 0.f : 1.f;
#pragma unroll
    for (int c = 0; c < 16; ++c) {
        int col = lane + 32 * c;
        x[(size_t)row * DD + col] =
            ((v[c] - mean) * inv * g[col] + beta[col]) * w;
    }
}

// ---------------------------------------------------------------------------
// Fused attention for DH=8: scores = q.kT + bias (written back in place as the
// evolving pair bias), softmax with -inf masking, A@V -> o (f16). One block
// per (b,h); k/v head slab cached in LDS; one wave per query row.
// ---------------------------------------------------------------------------
__launch_bounds__(256)
__global__ void attention_kernel(const _Float16* __restrict__ q,
                                 const _Float16* __restrict__ k,
                                 const _Float16* __restrict__ v,
                                 float* __restrict__ bias,
                                 _Float16* __restrict__ o)
{
    __shared__ __align__(16) _Float16 sk[LL * DHH];
    __shared__ __align__(16) _Float16 sv[LL * DHH];
    const int bh = blockIdx.x;
    const int b = bh / HH, h = bh % HH;
    const int t = threadIdx.x;
    { // stage K,V for this head (4 KB each)
        size_t src = ((size_t)(b * LL + t)) * DD + h * DHH;
        *(v8h*)(sk + t * DHH) = *(const v8h*)(k + src);
        *(v8h*)(sv + t * DHH) = *(const v8h*)(v + src);
    }
    __syncthreads();

    const int wid = t >> 5, lane = t & 31;
    for (int r = 0; r < 32; ++r) {
        const int i = wid * 32 + r;
        float qf[8];
        {
            v8h qv = *(const v8h*)(q + ((size_t)(b * LL + i)) * DD + h * DHH);
#pragma unroll
            for (int d = 0; d < 8; ++d) qf[d] = (float)qv[d];
        }
        float* brow = bias + ((size_t)(b * HH + h) * LL + i) * LL;
        float s[8], mx = -__builtin_inff();
#pragma unroll
        for (int jj = 0; jj < 8; ++jj) {
            int j = lane + 32 * jj;
            const _Float16* kr = sk + j * DHH;
            float dot = 0.f;
#pragma unroll
            for (int d = 0; d < 8; ++d) dot += qf[d] * (float)kr[d];
            float sc = dot + brow[j];
            brow[j] = sc;                 // evolving pre-softmax bias
            s[jj] = sc;
            mx = fmaxf(mx, sc);
        }
        mx = wave_max(mx);
        float sum = 0.f, of[8] = {0, 0, 0, 0, 0, 0, 0, 0};
#pragma unroll
        for (int jj = 0; jj < 8; ++jj) {
            float pex = (mx == -__builtin_inff()) ? 0.f : __expf(s[jj] - mx);
            sum += pex;
            const _Float16* vr = sv + (lane + 32 * jj) * DHH;
#pragma unroll
            for (int d = 0; d < 8; ++d) of[d] += pex * (float)vr[d];
        }
        sum = wave_sum(sum);
#pragma unroll
        for (int d = 0; d < 8; ++d) of[d] = wave_sum(of[d]);
        if (lane == 0) {
            float invs = (sum > 0.f) ? (1.0f / sum) : 0.f;
            v8h ov;
#pragma unroll
            for (int d = 0; d < 8; ++d) ov[d] = (_Float16)(of[d] * invs);
            *(v8h*)(o + ((size_t)(b * LL + i)) * DD + h * DHH) = ov;
        }
    }
}

// ---------------------------------------------------------------------------
// Final LN + pad mask + mean over L -> out (B, D)
// ---------------------------------------------------------------------------
__launch_bounds__(256)
__global__ void final_kernel(const float* __restrict__ x,
                             const int* __restrict__ tokens,
                             const float* __restrict__ g,
                             const float* __restrict__ beta,
                             float* __restrict__ out)
{
    __shared__ float sAcc[8][DD];
    const int wid = threadIdx.x >> 5, lane = threadIdx.x & 31;
    const int b = blockIdx.x;
    float acc[16];
#pragma unroll
    for (int c = 0; c < 16; ++c) acc[c] = 0.f;

    for (int r = 0; r < 32; ++r) {
        const int i = wid * 32 + r;
        const float* xr = x + ((size_t)(b * LL + i)) * DD;
        float v[16], s = 0.f, sq = 0.f;
#pragma unroll
        for (int c = 0; c < 16; ++c) {
            v[c] = xr[lane + 32 * c];
            s += v[c]; sq += v[c] * v[c];
        }
        s = wave_sum(s); sq = wave_sum(sq);
        const float mean = s * (1.0f / DD);
        const float inv  = rsqrtf(sq * (1.0f / DD) - mean * mean + 1e-5f);
        const float w = (tokens[b * LL + i] == 0) ? 0.f : (1.0f / LL);
#pragma unroll
        for (int c = 0; c < 16; ++c) {
            int col = lane + 32 * c;
            acc[c] += ((v[c] - mean) * inv * g[col] + beta[col]) * w;
        }
    }
#pragma unroll
    for (int c = 0; c < 16; ++c) sAcc[wid][lane + 32 * c] = acc[c];
    __syncthreads();
    for (int col = threadIdx.x; col < DD; col += 256) {
        float s = 0.f;
#pragma unroll
        for (int w = 0; w < 8; ++w) s += sAcc[w][col];
        out[(size_t)b * DD + col] = s;
    }
}

// ---------------------------------------------------------------------------
// host launcher
// ---------------------------------------------------------------------------
extern "C" void kernel_launch(void* const* d_in, const int* in_sizes, int n_in,
                              void* d_out, int out_size, void* d_ws, size_t ws_size,
                              hipStream_t stream)
{
    (void)in_sizes; (void)n_in; (void)out_size; (void)ws_size;
    const int*   src_tokens    = (const int*)  d_in[0];
    const float* src_distance  = (const float*)d_in[1];
    const int*   src_edge_type = (const int*)  d_in[2];
    const float* embed_tokens  = (const float*)d_in[3];
    const float* gbf_means     = (const float*)d_in[4];
    const float* gbf_stds      = (const float*)d_in[5];
    const float* gbf_mul       = (const float*)d_in[6];
    const float* gbf_bias_in   = (const float*)d_in[7];
    const float* proj_w1       = (const float*)d_in[8];
    const float* proj_b1       = (const float*)d_in[9];
    const float* proj_w2       = (const float*)d_in[10];
    const float* proj_b2       = (const float*)d_in[11];
    const float* emb_ln_g      = (const float*)d_in[12];
    const float* emb_ln_b      = (const float*)d_in[13];
    const float* ln1_g         = (const float*)d_in[14];
    const float* ln1_b         = (const float*)d_in[15];
    const float* wq            = (const float*)d_in[16];
    const float* bq            = (const float*)d_in[17];
    const float* wk            = (const float*)d_in[18];
    const float* bk            = (const float*)d_in[19];
    const float* wv            = (const float*)d_in[20];
    const float* bv            = (const float*)d_in[21];
    const float* wo            = (const float*)d_in[22];
    const float* bo            = (const float*)d_in[23];
    const float* ln2_g         = (const float*)d_in[24];
    const float* ln2_b         = (const float*)d_in[25];
    const float* wf1           = (const float*)d_in[26];
    const float* bf1           = (const float*)d_in[27];
    const float* wf2           = (const float*)d_in[28];
    const float* bf2           = (const float*)d_in[29];
    const float* final_ln_g    = (const float*)d_in[30];
    const float* final_ln_b    = (const float*)d_in[31];

    // workspace carve-out (256B aligned)
    char* ws = (char*)d_ws;
    size_t off = 0;
    auto carve = [&](size_t bytes) {
        void* p = ws + off;
        off += (bytes + 255) & ~(size_t)255;
        return p;
    };
    float*    x    = (float*)   carve((size_t)MROWS * DD * 4);
    _Float16* h16  = (_Float16*)carve((size_t)MROWS * DD * 2);
    _Float16* q16  = (_Float16*)carve((size_t)MROWS * DD * 2);
    _Float16* k16  = (_Float16*)carve((size_t)MROWS * DD * 2);
    _Float16* v16  = (_Float16*)carve((size_t)MROWS * DD * 2);
    _Float16* o16  = (_Float16*)carve((size_t)MROWS * DD * 2);
    _Float16* t16  = (_Float16*)carve((size_t)MROWS * FF * 2);
    float*    biasb = (float*)  carve((size_t)BB * HH * LL * LL * 4);

    // 1. fused pair-bias head (WMMA)
    gbf_bias_kernel<<<NPAIR / 64, 128, 0, stream>>>(
        src_tokens, src_distance, src_edge_type,
        gbf_means, gbf_stds, gbf_mul, gbf_bias_in,
        proj_w1, proj_b1, proj_w2, proj_b2, biasb);

    // 2. embedding + embed LN
    embed_ln_kernel<<<MROWS / 8, 256, 0, stream>>>(
        src_tokens, embed_tokens, emb_ln_g, emb_ln_b, x);

    const float qscale = 0.35355339059327373f;  // DH^-0.5
    const dim3 gD(DD / 128, MROWS / 32);   // N=512 GEMMs
    const dim3 gF(FF / 128, MROWS / 32);   // N=2048 GEMM

    for (int l = 0; l < NLAYER; ++l) {
        const float* wq_l = wq + (size_t)l * DD * DD;
        const float* wk_l = wk + (size_t)l * DD * DD;
        const float* wv_l = wv + (size_t)l * DD * DD;
        const float* wo_l = wo + (size_t)l * DD * DD;
        const float* wf1_l = wf1 + (size_t)l * DD * FF;
        const float* wf2_l = wf2 + (size_t)l * FF * DD;

        // pre-LN -> f16
        ln_to_half_kernel<<<MROWS / 8, 256, 0, stream>>>(
            x, ln1_g + l * DD, ln1_b + l * DD, h16);
        // QKV projections (WMMA); q pre-scaled
        gemm_kernel<0, 0, 1><<<gD, 128, 0, stream>>>(
            h16, wq_l, bq + l * DD, nullptr, q16, MROWS, DD, DD, qscale);
        gemm_kernel<0, 0, 1><<<gD, 128, 0, stream>>>(
            h16, wk_l, bk + l * DD, nullptr, k16, MROWS, DD, DD, 1.0f);
        gemm_kernel<0, 0, 1><<<gD, 128, 0, stream>>>(
            h16, wv_l, bv + l * DD, nullptr, v16, MROWS, DD, DD, 1.0f);
        // fused attention; bias updated in place to the new pre-softmax scores
        attention_kernel<<<BB * HH, 256, 0, stream>>>(q16, k16, v16, biasb, o16);
        // output projection + residual (fp32 stream)
        gemm_kernel<0, 1, 0><<<gD, 128, 0, stream>>>(
            o16, wo_l, bo + l * DD, x, x, MROWS, DD, DD, 1.0f);
        // FFN
        ln_to_half_kernel<<<MROWS / 8, 256, 0, stream>>>(
            x, ln2_g + l * DD, ln2_b + l * DD, h16);
        gemm_kernel<1, 0, 1><<<gF, 128, 0, stream>>>(
            h16, wf1_l, bf1 + l * FF, nullptr, t16, MROWS, FF, DD, 1.0f);
        gemm_kernel<0, 1, 0><<<gD, 128, 0, stream>>>(
            t16, wf2_l, bf2 + l * DD, x, x, MROWS, DD, FF, 1.0f);
    }

    // final LN + mask + mean over sequence
    final_kernel<<<BB, 256, 0, stream>>>(
        x, src_tokens, final_ln_g, final_ln_b, (float*)d_out);
}
